// TalkerPrefillForExport_66864050864550
// MI455X (gfx1250) — matrix operand order repaired and checked by reading, not compiled, not run on hardware
//
#include <hip/hip_runtime.h>

// ---------------- model constants ----------------
constexpr int L_ = 28, H_ = 1024, NH_ = 16, NKV_ = 8, HD_ = 128;
constexpr int FF_ = 3072, V_ = 3072, T_ = 1024;
constexpr float EPS_ = 1e-6f;

// ---------------- WMMA types ----------------
typedef __attribute__((ext_vector_type(16))) _Float16 v16h;
typedef __attribute__((ext_vector_type(8)))  float    v8f;

union HFrag { v16h v; unsigned int u[8]; };

__device__ __forceinline__ unsigned short f2h_(float f) {
  union { _Float16 h; unsigned short u; } c;
  c.h = (_Float16)f;
  return c.u;
}

// =========================================================================
// Generic batched GEMM: C[M x N] (+)= A[M x K] * B, fp32 in/out, fp16 WMMA
// with fp32 accumulation.  M == T_ always (grid.y = T_/128).
//   BT==0: B is K x N row-major (ldb = row stride)
//   BT==1: B source is N x K row-major (ldb = row stride) -> B = src^T
//   EPI: 0 = store, 1 = C += acc, 2 = store acc*scale with causal mask
//   batch b: A += b*sA ; B += (b>>bShift)*sB ; C += b*sC
// Software pipeline: next tile's global loads are issued into registers
// before computing the current LDS tile (2-barrier single-LDS-buffer).
// =========================================================================
constexpr int LDSS = 72;  // LDS stride in halves; 144B rows -> conflict-free frag reads

template <int BT, int EPI>
__global__ __launch_bounds__(256) void gemm_f16_kernel(
    const float* __restrict__ A, const float* __restrict__ B, float* __restrict__ C,
    int N, int K, int lda, int ldb, int ldc,
    long sA, long sB, long sC, int bShift, float scale)
{
  __shared__ unsigned short Al[128 * LDSS];
  __shared__ unsigned short Bl[128 * LDSS];

  const int tid  = threadIdx.x;
  const int lane = tid & 31;
  const int w    = tid >> 5;
  const int wm   = w >> 2;       // 0..1
  const int wn   = w & 3;        // 0..3
  const int hlf  = lane >> 4;    // 0..1
  const int l15  = lane & 15;

  const int mBase = blockIdx.y * 128;
  const int nBase = blockIdx.x * 128;
  const int b     = blockIdx.z;

  A += (size_t)b * sA;
  B += (size_t)(b >> bShift) * sB;
  C += (size_t)b * sC;

  float4 aReg[8], bReg[8];

  const int rA = tid >> 4;                // 0..15
  const int cA = (tid & 15) << 2;         // 0..60
  const int rB = tid >> 5;                // 0..7   (BT==0)
  const int cB = (tid & 31) << 2;         // 0..124 (BT==0)

  auto loadTiles = [&](int k0) {
#pragma unroll
    for (int p = 0; p < 8; ++p)
      aReg[p] = *(const float4*)(A + (size_t)(mBase + rA + p * 16) * lda + k0 + cA);
    if (BT) {
#pragma unroll
      for (int p = 0; p < 8; ++p)
        bReg[p] = *(const float4*)(B + (size_t)(nBase + rA + p * 16) * ldb + k0 + cA);
    } else {
#pragma unroll
      for (int p = 0; p < 8; ++p)
        bReg[p] = *(const float4*)(B + (size_t)(k0 + rB + p * 8) * ldb + nBase + cB);
    }
  };

  auto storeTiles = [&]() {
#pragma unroll
    for (int p = 0; p < 8; ++p) {
      const float4 f = aReg[p];
      const unsigned int lo = (unsigned)f2h_(f.x) | ((unsigned)f2h_(f.y) << 16);
      const unsigned int hi = (unsigned)f2h_(f.z) | ((unsigned)f2h_(f.w) << 16);
      *(uint2*)&Al[(rA + p * 16) * LDSS + cA] = make_uint2(lo, hi);
    }
    if (BT) {
#pragma unroll
      for (int p = 0; p < 8; ++p) {
        const float4 f = bReg[p];
        const unsigned int lo = (unsigned)f2h_(f.x) | ((unsigned)f2h_(f.y) << 16);
        const unsigned int hi = (unsigned)f2h_(f.z) | ((unsigned)f2h_(f.w) << 16);
        *(uint2*)&Bl[(rA + p * 16) * LDSS + cA] = make_uint2(lo, hi);
      }
    } else {
#pragma unroll
      for (int p = 0; p < 8; ++p) {
        const float4 f = bReg[p];
        const int k = rB + p * 8;
        Bl[(cB + 0) * LDSS + k] = f2h_(f.x);
        Bl[(cB + 1) * LDSS + k] = f2h_(f.y);
        Bl[(cB + 2) * LDSS + k] = f2h_(f.z);
        Bl[(cB + 3) * LDSS + k] = f2h_(f.w);
      }
    }
  };

  v8f acc[4][2];
#pragma unroll
  for (int i = 0; i < 4; ++i)
#pragma unroll
    for (int j = 0; j < 2; ++j)
#pragma unroll
      for (int e = 0; e < 8; ++e) acc[i][j][e] = 0.0f;

  const int nT = K >> 6;  // K / 64
  loadTiles(0);
  storeTiles();

  for (int kt = 0; kt < nT; ++kt) {
    __syncthreads();                       // staged tile visible to all
    if (kt + 1 < nT) loadTiles((kt + 1) << 6);  // in flight during compute

#pragma unroll
    for (int kk = 0; kk < 64; kk += 32) {
      HFrag af[4], bf[2];
#pragma unroll
      for (int i = 0; i < 4; ++i) {
        const int m = wm * 64 + i * 16 + l15;
        const unsigned short* row = &Al[m * LDSS + kk];
#pragma unroll
        for (int j = 0; j < 8; ++j) {
          const int k = ((j < 4) ? 0 : 16) + hlf * 8 + ((j & 3) << 1);
          af[i].u[j] = *(const unsigned int*)(row + k);
        }
      }
#pragma unroll
      for (int j2 = 0; j2 < 2; ++j2) {
        const int n = wn * 32 + j2 * 16 + l15;
        const unsigned short* row = &Bl[n * LDSS + kk];
#pragma unroll
        for (int j = 0; j < 8; ++j) {
          const int k = ((j < 4) ? 0 : 16) + hlf * 8 + ((j & 3) << 1);
          bf[j2].u[j] = *(const unsigned int*)(row + k);
        }
      }
#pragma unroll
      for (int i = 0; i < 4; ++i)
#pragma unroll
        for (int j = 0; j < 2; ++j)
          acc[i][j] = __builtin_amdgcn_wmma_f32_16x16x32_f16(
              false, af[i].v, false, bf[j].v, (short)0, acc[i][j], false, false);
    }

    __syncthreads();                       // all reads of tile done
    if (kt + 1 < nT) storeTiles();         // overwrite LDS with next tile
  }

  // ---- epilogue ----
#pragma unroll
  for (int i = 0; i < 4; ++i)
#pragma unroll
    for (int j = 0; j < 2; ++j) {
      const int gm0 = mBase + wm * 64 + i * 16 + hlf * 8;
      const int gn  = nBase + wn * 32 + j * 16 + l15;
#pragma unroll
      for (int r = 0; r < 8; ++r) {
        const int gm = gm0 + r;
        const float v = acc[i][j][r];
        const size_t idx = (size_t)gm * ldc + gn;
        if (EPI == 1)       C[idx] += v;
        else if (EPI == 2)  C[idx] = (gn > gm) ? -3.0e38f : v * scale;
        else                C[idx] = v;
      }
    }
}

// =========================================================================
// RMSNorm over rows of length H_:  y = x * rsqrt(mean(x^2)+eps) * w
// =========================================================================
__global__ __launch_bounds__(256) void rms_kernel(const float* __restrict__ x,
                                                  const float* __restrict__ w,
                                                  float* __restrict__ y) {
  const int row = blockIdx.x;
  const int tid = threadIdx.x;
  const float* xr = x + (size_t)row * H_;
  float* yr = y + (size_t)row * H_;
  float v[4]; float s = 0.f;
#pragma unroll
  for (int i = 0; i < 4; ++i) { v[i] = xr[tid + i * 256]; s += v[i] * v[i]; }
  __shared__ float red[256];
  red[tid] = s; __syncthreads();
  for (int st = 128; st > 0; st >>= 1) { if (tid < st) red[tid] += red[tid + st]; __syncthreads(); }
  const float inv = rsqrtf(red[0] * (1.0f / H_) + EPS_);
#pragma unroll
  for (int i = 0; i < 4; ++i) yr[tid + i * 256] = v[i] * inv * w[tid + i * 256];
}

// =========================================================================
// mRoPE cos/sin table: (T, HD) with section select (16,24,24)*2
// =========================================================================
__global__ __launch_bounds__(128) void rope_table_kernel(const int* __restrict__ pid,
                                                         float* __restrict__ cosb,
                                                         float* __restrict__ sinb) {
  const int t = blockIdx.x, j = threadIdx.x;
  const int sec = (j < 32) ? 0 : ((j < 80) ? 1 : 2);
  const float pos = (float)pid[sec * T_ + t];
  const int i = j & 63;
  const float invf = powf(1000000.0f, -((float)(2 * i)) / 128.0f);
  const float a = pos * invf;
  cosb[t * HD_ + j] = cosf(a);
  sinb[t * HD_ + j] = sinf(a);
}

// =========================================================================
// Per-head RMSNorm + RoPE for q/k; writes roped k and v into d_out
// keys/vals (layout NKV x T x HD).  grid = (NH_+NKV_, T_), block = HD_
// =========================================================================
__global__ __launch_bounds__(128) void qk_rope_kernel(float* __restrict__ qb,
                                                      float* __restrict__ kb,
                                                      const float* __restrict__ vb,
                                                      const float* __restrict__ qnw,
                                                      const float* __restrict__ knw,
                                                      const float* __restrict__ cosb,
                                                      const float* __restrict__ sinb,
                                                      float* __restrict__ keysL,
                                                      float* __restrict__ valsL) {
  const int hh = blockIdx.x, t = blockIdx.y, d = threadIdx.x;
  const bool isq = hh < NH_;
  const int kh = hh - NH_;
  float* src;
  const float* w;
  if (isq) { src = qb + ((size_t)t * NH_ + hh) * HD_;  w = qnw; }
  else     { src = kb + ((size_t)t * NKV_ + kh) * HD_; w = knw; }
  const float x = src[d];
  __shared__ float sh[128];
  __shared__ float sy[128];
  sh[d] = x * x; __syncthreads();
  for (int st = 64; st > 0; st >>= 1) { if (d < st) sh[d] += sh[d + st]; __syncthreads(); }
  const float y = x * rsqrtf(sh[0] * (1.0f / HD_) + EPS_) * w[d];
  sy[d] = y; __syncthreads();
  const float rot = (d < 64) ? -sy[d + 64] : sy[d - 64];
  const float o = y * cosb[t * HD_ + d] + rot * sinb[t * HD_ + d];
  if (isq) {
    src[d] = o;
  } else {
    keysL[((size_t)kh * T_ + t) * HD_ + d] = o;
    valsL[((size_t)kh * T_ + t) * HD_ + d] = vb[((size_t)t * NKV_ + kh) * HD_ + d];
  }
}

// =========================================================================
// Row softmax over length T_ (rows = NH_*T_)
// =========================================================================
__global__ __launch_bounds__(256) void softmax_kernel(float* __restrict__ sc) {
  const int row = blockIdx.x, tid = threadIdx.x;
  float* p = sc + (size_t)row * T_;
  float v[4]; float mx = -3.4e38f;
#pragma unroll
  for (int i = 0; i < 4; ++i) { v[i] = p[tid + i * 256]; mx = fmaxf(mx, v[i]); }
  __shared__ float red[256];
  red[tid] = mx; __syncthreads();
  for (int st = 128; st > 0; st >>= 1) { if (tid < st) red[tid] = fmaxf(red[tid], red[tid + st]); __syncthreads(); }
  mx = red[0]; __syncthreads();
  float s = 0.f;
#pragma unroll
  for (int i = 0; i < 4; ++i) { v[i] = expf(v[i] - mx); s += v[i]; }
  red[tid] = s; __syncthreads();
  for (int st = 128; st > 0; st >>= 1) { if (tid < st) red[tid] += red[tid + st]; __syncthreads(); }
  const float inv = 1.0f / red[0];
#pragma unroll
  for (int i = 0; i < 4; ++i) p[tid + i * 256] = v[i] * inv;
}

// =========================================================================
// g = silu(g) * u, elementwise (float4) over T_*FF_
// =========================================================================
__global__ __launch_bounds__(256) void silu_mul_kernel(float* __restrict__ g,
                                                       const float* __restrict__ u) {
  const size_t i = ((size_t)blockIdx.x * 256 + threadIdx.x) * 4;
  float4 x = *(float4*)(g + i);
  const float4 y = *(const float4*)(u + i);
  x.x = (x.x / (1.0f + expf(-x.x))) * y.x;
  x.y = (x.y / (1.0f + expf(-x.y))) * y.y;
  x.z = (x.z / (1.0f + expf(-x.z))) * y.z;
  x.w = (x.w / (1.0f + expf(-x.w))) * y.w;
  *(float4*)(g + i) = x;
}

// ---------------- host-side launch helpers ----------------
static void gemm(hipStream_t s, const float* A, const float* B, float* C,
                 int M, int N, int K, int lda, int ldb, int ldc,
                 long sA, long sB, long sC, int batch, int bShift,
                 int bTrans, int epi, float scale) {
  dim3 grid(N / 128, M / 128, batch);
#define GEMM_LAUNCH(bt, ep)                                                     \
  gemm_f16_kernel<bt, ep><<<grid, 256, 0, s>>>(A, B, C, N, K, lda, ldb, ldc,    \
                                               sA, sB, sC, bShift, scale)
  if (bTrans) {
    if (epi == 2)      GEMM_LAUNCH(1, 2);
    else if (epi == 1) GEMM_LAUNCH(1, 1);
    else               GEMM_LAUNCH(1, 0);
  } else {
    if (epi == 2)      GEMM_LAUNCH(0, 2);
    else if (epi == 1) GEMM_LAUNCH(0, 1);
    else               GEMM_LAUNCH(0, 0);
  }
#undef GEMM_LAUNCH
}

extern "C" void kernel_launch(void* const* d_in, const int* in_sizes, int n_in,
                              void* d_out, int out_size, void* d_ws, size_t ws_size,
                              hipStream_t stream) {
  (void)in_sizes; (void)n_in; (void)out_size; (void)ws_size;
  const float* embeds = (const float*)d_in[0];
  const int*   posids = (const int*)d_in[1];
  const float* Wq  = (const float*)d_in[2];
  const float* Wk  = (const float*)d_in[3];
  const float* Wv  = (const float*)d_in[4];
  const float* Wo  = (const float*)d_in[5];
  const float* qnw = (const float*)d_in[6];
  const float* knw = (const float*)d_in[7];
  const float* ln1 = (const float*)d_in[8];
  const float* ln2 = (const float*)d_in[9];
  const float* gw  = (const float*)d_in[10];
  const float* uw  = (const float*)d_in[11];
  const float* dwn = (const float*)d_in[12];
  const float* nw  = (const float*)d_in[13];
  const float* hw  = (const float*)d_in[14];

  float* out    = (float*)d_out;
  float* logits = out;
  float* hout   = logits + (size_t)T_ * V_;
  float* keys0  = hout + (size_t)T_ * H_;
  float* vals0  = keys0 + (size_t)L_ * NKV_ * T_ * HD_;

  float* ws = (float*)d_ws;
  size_t off = 0;
  auto alloc = [&](size_t n) { float* p = ws + off; off += n; return p; };
  float* hcur = alloc((size_t)T_ * H_);
  float* xn   = alloc((size_t)T_ * H_);
  float* qb   = alloc((size_t)T_ * NH_ * HD_);
  float* kb   = alloc((size_t)T_ * NKV_ * HD_);
  float* vb   = alloc((size_t)T_ * NKV_ * HD_);
  float* ao   = alloc((size_t)T_ * NH_ * HD_);
  float* sc   = alloc((size_t)NH_ * T_ * T_);
  float* gb   = alloc((size_t)T_ * FF_);
  float* ub   = alloc((size_t)T_ * FF_);
  float* cosb = alloc((size_t)T_ * HD_);
  float* sinb = alloc((size_t)T_ * HD_);

  hipMemcpyAsync(hcur, embeds, sizeof(float) * (size_t)T_ * H_,
                 hipMemcpyDeviceToDevice, stream);
  rope_table_kernel<<<T_, HD_, 0, stream>>>(posids, cosb, sinb);

  const float attnScale = 0.08838834764831845f;  // 1/sqrt(128)
  const int QD = NH_ * HD_;    // 2048
  const int KD = NKV_ * HD_;   // 1024

  for (int l = 0; l < L_; ++l) {
    const float* wq = Wq + (size_t)l * H_ * QD;
    const float* wk = Wk + (size_t)l * H_ * KD;
    const float* wv = Wv + (size_t)l * H_ * KD;
    const float* wo = Wo + (size_t)l * QD * H_;
    const float* g_ = gw + (size_t)l * H_ * FF_;
    const float* u_ = uw + (size_t)l * H_ * FF_;
    const float* d_ = dwn + (size_t)l * FF_ * H_;
    float* keysL = keys0 + (size_t)l * NKV_ * T_ * HD_;
    float* valsL = vals0 + (size_t)l * NKV_ * T_ * HD_;

    // x = rms(h, ln1)
    rms_kernel<<<T_, 256, 0, stream>>>(hcur, ln1 + (size_t)l * H_, xn);
    // q,k,v projections
    gemm(stream, xn, wq, qb, T_, QD, H_, H_, QD, QD, 0, 0, 0, 1, 0, 0, 0, 0.f);
    gemm(stream, xn, wk, kb, T_, KD, H_, H_, KD, KD, 0, 0, 0, 1, 0, 0, 0, 0.f);
    gemm(stream, xn, wv, vb, T_, KD, H_, H_, KD, KD, 0, 0, 0, 1, 0, 0, 0, 0.f);
    // per-head rmsnorm + rope; roped k and v -> d_out kv cache
    qk_rope_kernel<<<dim3(NH_ + NKV_, T_), HD_, 0, stream>>>(
        qb, kb, vb, qnw + (size_t)l * HD_, knw + (size_t)l * HD_,
        cosb, sinb, keysL, valsL);
    // scores = q @ k^T * scale + causal mask   (batched over NH_, GQA shift 1)
    gemm(stream, qb, keysL, sc, T_, T_, HD_, QD, HD_, T_,
         HD_, (long)T_ * HD_, (long)T_ * T_, NH_, 1, 1, 2, attnScale);
    softmax_kernel<<<NH_ * T_, 256, 0, stream>>>(sc);
    // attn out = P @ V
    gemm(stream, sc, valsL, ao, T_, HD_, T_, T_, HD_, QD,
         (long)T_ * T_, (long)T_ * HD_, HD_, NH_, 1, 0, 0, 0.f);
    // h += ao @ Wo
    gemm(stream, ao, wo, hcur, T_, H_, QD, QD, H_, H_, 0, 0, 0, 1, 0, 0, 1, 0.f);
    // x2 = rms(h, ln2); mlp
    rms_kernel<<<T_, 256, 0, stream>>>(hcur, ln2 + (size_t)l * H_, xn);
    gemm(stream, xn, g_, gb, T_, FF_, H_, H_, FF_, FF_, 0, 0, 0, 1, 0, 0, 0, 0.f);
    gemm(stream, xn, u_, ub, T_, FF_, H_, H_, FF_, FF_, 0, 0, 0, 1, 0, 0, 0, 0.f);
    silu_mul_kernel<<<(T_ * FF_) / 1024, 256, 0, stream>>>(gb, ub);
    gemm(stream, gb, d_, hcur, T_, H_, FF_, FF_, H_, H_, 0, 0, 0, 1, 0, 0, 1, 0.f);
  }

  // final norm -> h output, then logits
  rms_kernel<<<T_, 256, 0, stream>>>(hcur, nw, hout);
  gemm(stream, hout, hw, logits, T_, V_, H_, H_, V_, V_, 0, 0, 0, 1, 0, 0, 0, 0.f);
}
